// _AxisSlidingAttention_22093311771366
// MI455X (gfx1250) — matrix operand order repaired and checked by reading, not compiled, not run on hardware
//
#include <hip/hip_runtime.h>
#include <hip/hip_bf16.h>
#include <math.h>

// ---------------- problem constants ----------------
#define BB 2
#define TT 1024
#define EE 1024
#define HH 8
#define HD 128
#define WS 64
#define TAU 10.0f
#define SCALEF 0.08838834764831845f   // 1/sqrt(128)
#define NEGF  -3.4028234663852886e38f // finfo(f32).min

#define MT (BB * TT)                  // 2048 GEMM rows
#define NSUB 4                        // 16x16 tiles per wave in N

typedef __attribute__((ext_vector_type(16))) __bf16 v16bf;
typedef __attribute__((ext_vector_type(8)))  float  v8f;

// ---------------- helpers ----------------
__device__ __forceinline__ __bf16 f2bf(float x) {
    unsigned int u = __float_as_uint(x);
    u += 0x7fffu + ((u >> 16) & 1u);          // round-to-nearest-even
    unsigned short s = (unsigned short)(u >> 16);
    __bf16 r;
    __builtin_memcpy(&r, &s, 2);
    return r;
}

// A fragment: 16x32 (MxK) bf16, source row-major with leading dim ld.
// ISA layout: lanes 0-15 -> M=lane, halfs 0..7 = K 0..7, halfs 8..15 = K 16..23
//             lanes 16-31 -> M=lane-16, halfs 0..7 = K 8..15, halfs 8..15 = K 24..31
__device__ __forceinline__ v16bf load_a_frag(const __bf16* base, int ld,
                                             int row0, int k0, int lane) {
    const int m  = lane & 15;
    const int hi = lane >> 4;
    const __bf16* p = base + (size_t)(row0 + m) * ld + k0 + hi * 8;
    v16bf f;
#pragma unroll
    for (int e = 0; e < 8; ++e) { f[e] = p[e]; f[e + 8] = p[e + 16]; }
    return f;
}

// B fragment: 32x16 (KxN) where B[k][n] = Wt[n][k], Wt row-major N x K.
// ISA layout: lanes 0-15 -> N=lane, halfs 0..7 = K 0..7, halfs 8..15 = K 16..23; lanes 16-31 -> N=lane-16, K 8..15, 24..31.
__device__ __forceinline__ v16bf load_b_frag(const __bf16* base, int ld,
                                             int n0, int k0, int lane) {
    const int n  = lane & 15;
    const int kb = (lane >> 4) * 8;
    const __bf16* p = base + (size_t)(n0 + n) * ld + k0 + kb;
    v16bf f;
#pragma unroll
    for (int e = 0; e < 8; ++e) { f[e] = p[e]; f[e + 8] = p[e + 16]; }
    return f;
}

// ---------------- kernel 0: per-batch bitonic argsort ----------------
__global__ void sort_kernel(const float* __restrict__ coord,
                            int* __restrict__ sort_idx,
                            int* __restrict__ rowmap,
                            float* __restrict__ c_s,
                            int* __restrict__ depot) {
    const int b = blockIdx.x;
    const int t = threadIdx.x;
    __shared__ float skey[TT];
    __shared__ int   sidx[TT];
    skey[t] = coord[b * TT + t];
    sidx[t] = t;
    __syncthreads();
    for (int ksz = 2; ksz <= TT; ksz <<= 1) {
        for (int j = ksz >> 1; j > 0; j >>= 1) {
            const int ixj = t ^ j;
            if (ixj > t) {
                const bool up = ((t & ksz) == 0);
                const float ka = skey[t], kb2 = skey[ixj];
                const int   ia = sidx[t], ib  = sidx[ixj];
                const bool gt = (ka > kb2) || (ka == kb2 && ia > ib); // stable
                if (gt == up) {
                    skey[t] = kb2; skey[ixj] = ka;
                    sidx[t] = ib;  sidx[ixj] = ia;
                }
            }
            __syncthreads();
        }
    }
    sort_idx[b * TT + t] = sidx[t];
    rowmap[b * TT + t]   = sidx[t];   // output scatter: out[b, sidx[t]] = sorted_out[b, t]
    c_s[b * TT + t]      = skey[t];
    if (sidx[t] == 0) depot[b] = t;
}

// ---------------- f32 -> bf16 convert (weights) ----------------
__global__ void conv_bf16_kernel(const float* __restrict__ in,
                                 __bf16* __restrict__ out, int n) {
    const int g = blockIdx.x * blockDim.x + threadIdx.x;
    if (g < n) out[g] = f2bf(in[g]);
}

// ---------------- gather h into sorted order + convert ----------------
__global__ void gather_h_kernel(const float* __restrict__ h,
                                const int* __restrict__ sort_idx,
                                __bf16* __restrict__ hs) {
    const int g = blockIdx.x * blockDim.x + threadIdx.x;
    if (g >= BB * TT * EE) return;
    const int e  = g % EE;
    const int bt = g / EE;
    const int b  = bt / TT;
    const int src = sort_idx[bt];
    hs[g] = f2bf(h[(size_t)(b * TT + src) * EE + e]);
}

// ---------------- WMMA GEMM: C[m,n] = sum_k A[m,k]*Wt[n,k] + bias[n] ----------------
// M=MT, N=K=EE. Each wave owns a 16 x (16*NSUB) strip; A fragment reused NSUB times.
// Optional rowmap scatters output rows within each batch (fused unsort).
__global__ void gemm_bf16_kernel(const __bf16* __restrict__ A,
                                 const __bf16* __restrict__ Wt,
                                 const float* __restrict__ bias,
                                 float* __restrict__ C,
                                 const int* __restrict__ rowmap) {
    const int lane = threadIdx.x & 31;
    const int wid  = blockIdx.x * (blockDim.x >> 5) + (threadIdx.x >> 5);
    const int tiles_n = EE / (16 * NSUB);          // 16
    const int tm  = wid / tiles_n;                 // 0..127
    const int tnb = (wid % tiles_n) * NSUB;

    v8f acc[NSUB];
#pragma unroll
    for (int s = 0; s < NSUB; ++s)
        acc[s] = (v8f){0.f, 0.f, 0.f, 0.f, 0.f, 0.f, 0.f, 0.f};

    for (int k0 = 0; k0 < EE; k0 += 32) {
        const v16bf a = load_a_frag(A, EE, tm * 16, k0, lane);
#pragma unroll
        for (int s = 0; s < NSUB; ++s) {
            const v16bf bfrag = load_b_frag(Wt, EE, (tnb + s) * 16, k0, lane);
            acc[s] = __builtin_amdgcn_wmma_f32_16x16x32_bf16(
                false, a, false, bfrag, (short)0, acc[s], false, false);
        }
    }

    const int nl = lane & 15;
    const int mi = (lane >> 4) * 8;
#pragma unroll
    for (int s = 0; s < NSUB; ++s) {
        const int n  = (tnb + s) * 16 + nl;
        const float bv = bias[n];
#pragma unroll
        for (int r = 0; r < 8; ++r) {
            const int m = tm * 16 + mi + r;
            const int srow = rowmap ? ((m / TT) * TT + rowmap[m]) : m;
            C[(size_t)srow * EE + n] = acc[s][r] + bv;
        }
    }
}

// ---------------- attention: one block (128 thr) per (b, t), loops 8 heads ----------
__global__ void attn_kernel(const float* __restrict__ Q,
                            const float* __restrict__ K,
                            const float* __restrict__ V,
                            const float* __restrict__ c_s,
                            const int* __restrict__ depot,
                            __bf16* __restrict__ ctx) {
    const int bt = blockIdx.x;            // 0..B*T-1 (sorted order)
    const int b  = bt / TT;
    const int ts = bt % TT;
    const int d  = threadIdx.x;           // 0..127 (head dim)

    __shared__ float sQ[HD];
    __shared__ float sS[TT];              // up to full-T scores for the depot row

    const int   dep = depot[b];
    const float ct  = c_s[b * TT + ts];
    const bool  isDepot = (ts == dep);

    int start = ts - WS / 2;
    if (start < 0) start = 0;
    if (start > TT - WS) start = TT - WS;
    const bool hasDep = (dep >= start && dep < start + WS);
    const int  nw = isDepot ? TT : (WS + 1);

    for (int hh = 0; hh < HH; ++hh) {
        const int off = hh * HD;
        sQ[d] = Q[(size_t)bt * EE + off + d];
        __syncthreads();

        // scores
        for (int w = d; w < nw; w += HD) {
            const int kr = isDepot ? w : ((w < WS) ? (start + w) : dep);
            const float* kp = K + (size_t)(b * TT + kr) * EE + off;
            float dot = 0.f;
#pragma unroll 8
            for (int e = 0; e < HD; ++e) dot += sQ[e] * kp[e];
            const float cw = c_s[b * TT + kr];
            const float dd = cw - ct;
            float sc = dot * SCALEF - dd * dd * (1.0f / TAU);
            if (!isDepot && w == WS && hasDep) sc = NEGF;   // keep-mask on depot column
            sS[w] = sc;
        }
        __syncthreads();

        // softmax stats (redundant per thread; nw <= 1024)
        float mmax = NEGF;
        for (int w = 0; w < nw; ++w) mmax = fmaxf(mmax, sS[w]);
        float ssum = 0.f;
        for (int w = 0; w < nw; ++w) ssum += __expf(sS[w] - mmax);
        const float inv = 1.0f / ssum;

        // context for this lane's dim
        float acc = 0.f;
        for (int w = 0; w < nw; ++w) {
            const int kr = isDepot ? w : ((w < WS) ? (start + w) : dep);
            const float a = __expf(sS[w] - mmax) * inv;
            acc += a * V[(size_t)(b * TT + kr) * EE + off + d];
        }
        ctx[(size_t)bt * EE + off + d] = f2bf(acc);
        __syncthreads();
    }
}

// ---------------- host-side launcher ----------------
static inline size_t align256(size_t x) { return (x + 255) & ~(size_t)255; }

extern "C" void kernel_launch(void* const* d_in, const int* in_sizes, int n_in,
                              void* d_out, int out_size, void* d_ws, size_t ws_size,
                              hipStream_t stream) {
    const float* h      = (const float*)d_in[0];
    const float* coord  = (const float*)d_in[1];
    const float* Wq_w   = (const float*)d_in[2];
    const float* Wq_b   = (const float*)d_in[3];
    const float* Wk_w   = (const float*)d_in[4];
    const float* Wk_b   = (const float*)d_in[5];
    const float* Wv_w   = (const float*)d_in[6];
    const float* Wv_b   = (const float*)d_in[7];
    const float* Wo_w   = (const float*)d_in[8];
    const float* Wo_b   = (const float*)d_in[9];
    float* out = (float*)d_out;

    // ---- workspace carve-up ----
    char* w = (char*)d_ws;
    int*    sort_idx = (int*)w;    w += align256(sizeof(int) * MT);
    int*    rowmap   = (int*)w;    w += align256(sizeof(int) * MT);
    float*  c_s      = (float*)w;  w += align256(sizeof(float) * MT);
    int*    depot    = (int*)w;    w += align256(sizeof(int) * BB);
    __bf16* hs       = (__bf16*)w; w += align256(sizeof(__bf16) * (size_t)MT * EE);
    __bf16* wq       = (__bf16*)w; w += align256(sizeof(__bf16) * (size_t)EE * EE);
    __bf16* wk       = (__bf16*)w; w += align256(sizeof(__bf16) * (size_t)EE * EE);
    __bf16* wv       = (__bf16*)w; w += align256(sizeof(__bf16) * (size_t)EE * EE);
    __bf16* wo       = (__bf16*)w; w += align256(sizeof(__bf16) * (size_t)EE * EE);
    float*  Qb       = (float*)w;  w += align256(sizeof(float) * (size_t)MT * EE);
    float*  Kb       = (float*)w;  w += align256(sizeof(float) * (size_t)MT * EE);
    float*  Vb       = (float*)w;  w += align256(sizeof(float) * (size_t)MT * EE);
    __bf16* ctx      = (__bf16*)w; w += align256(sizeof(__bf16) * (size_t)MT * EE);
    (void)ws_size; (void)in_sizes; (void)n_in; (void)out_size;

    // 1) per-batch sort (1024-thread bitonic in LDS)
    sort_kernel<<<BB, TT, 0, stream>>>(coord, sort_idx, rowmap, c_s, depot);

    // 2) weight converts to bf16
    const int nW = EE * EE;
    const int cblk = 256, cgrd = (nW + cblk - 1) / cblk;
    conv_bf16_kernel<<<cgrd, cblk, 0, stream>>>(Wq_w, wq, nW);
    conv_bf16_kernel<<<cgrd, cblk, 0, stream>>>(Wk_w, wk, nW);
    conv_bf16_kernel<<<cgrd, cblk, 0, stream>>>(Wv_w, wv, nW);
    conv_bf16_kernel<<<cgrd, cblk, 0, stream>>>(Wo_w, wo, nW);

    // 3) gather h into sorted order (+ bf16)
    const int nH = MT * EE;
    gather_h_kernel<<<(nH + 255) / 256, 256, 0, stream>>>(h, sort_idx, hs);

    // 4) Q/K/V projections: WMMA bf16 GEMMs, fp32 accumulate
    const int gemm_grid = (MT / 16) * (EE / (16 * NSUB)) / 8;   // 8 waves/block
    gemm_bf16_kernel<<<gemm_grid, 256, 0, stream>>>(hs, wq, Wq_b, Qb, nullptr);
    gemm_bf16_kernel<<<gemm_grid, 256, 0, stream>>>(hs, wk, Wk_b, Kb, nullptr);
    gemm_bf16_kernel<<<gemm_grid, 256, 0, stream>>>(hs, wv, Wv_b, Vb, nullptr);

    // 5) sliding-window + depot attention (fp32 VALU; tiny FLOPs)
    attn_kernel<<<MT, HD, 0, stream>>>(Qb, Kb, Vb, c_s, depot, ctx);

    // 6) output projection with fused unsort (row scatter via sort_idx)
    gemm_bf16_kernel<<<gemm_grid, 256, 0, stream>>>(ctx, wo, Wo_b, out, rowmap);
}
